// Attention_90787018703157
// MI455X (gfx1250) — matrix operand order repaired and checked by reading, not compiled, not run on hardware
//
#include <hip/hip_runtime.h>

typedef __attribute__((ext_vector_type(16))) _Float16 v16h;
typedef __attribute__((ext_vector_type(8)))  float    v8f;
typedef __attribute__((ext_vector_type(4)))  float    v4f;

#define B_   8
#define N_   2048
#define D_   10
#define QK_  32

// ---------------------------------------------------------------------------
// Phase 1: q = s * Wq^T, k = s * Wk^T  -> f16 buffers in workspace.
// One thread per (b,n) row. W matrices staged in LDS (they are tiny).
// ---------------------------------------------------------------------------
__global__ void qk_project_kernel(const float* __restrict__ s,
                                  const float* __restrict__ Wq,
                                  const float* __restrict__ Wk,
                                  _Float16* __restrict__ qbuf,
                                  _Float16* __restrict__ kbuf) {
    __shared__ float wq_s[QK_ * D_];
    __shared__ float wk_s[QK_ * D_];
    const int t = threadIdx.x;
    for (int i = t; i < QK_ * D_; i += blockDim.x) {
        wq_s[i] = Wq[i];
        wk_s[i] = Wk[i];
    }
    __syncthreads();

    const size_t row = (size_t)blockIdx.x * blockDim.x + t;   // [0, B*N)
    const float* srow = s + row * D_;
    float sv[D_];
#pragma unroll
    for (int d = 0; d < D_; ++d) sv[d] = srow[d];

    _Float16* qrow = qbuf + row * QK_;
    _Float16* krow = kbuf + row * QK_;
#pragma unroll
    for (int j = 0; j < QK_; ++j) {
        float aq = 0.0f, ak = 0.0f;
#pragma unroll
        for (int d = 0; d < D_; ++d) {
            aq = fmaf(wq_s[j * D_ + d], sv[d], aq);
            ak = fmaf(wk_s[j * D_ + d], sv[d], ak);
        }
        qrow[j] = (_Float16)aq;
        krow[j] = (_Float16)ak;
    }
}

// ---------------------------------------------------------------------------
// Phase 2: one block = one 16-row stripe of one batch, all 2048 columns.
//   logits tile (16x16) via v_wmma_f32_16x16x32_f16 (K=QK_=32 in one op),
//   w = logits^2 * G fused at C-fragment granularity (NT streaming loads of G),
//   stripe buffered in LDS, row sums via shfl_xor + ds_add_f32,
//   normalized coalesced NT b128 writeout.
// ---------------------------------------------------------------------------
union ABFrag { v16h v; v4f f4[2]; };

__global__ void attn_row_kernel(const _Float16* __restrict__ qbuf,
                                const _Float16* __restrict__ kbuf,
                                const float* __restrict__ G,
                                float* __restrict__ out) {
    extern __shared__ float lds[];            // [16*N_] w stripe | [16] sums | [16] inv
    float* wtile  = lds;
    float* rowsum = lds + 16 * N_;
    float* inv    = rowsum + 16;

    const int tid  = threadIdx.x;
    const int lane = tid & 31;
    const int wave = tid >> 5;                // 0..7, owns columns [wave*256, wave*256+256)
    const int b    = blockIdx.x >> 7;         // 128 stripes per batch
    const int r0   = (blockIdx.x & 127) * 16; // stripe base row

    if (tid < 16) rowsum[tid] = 0.0f;
    __syncthreads();

    const int half = lane >> 4;               // 0: lanes 0-15, 1: lanes 16-31
    const int l16  = lane & 15;

    // A fragment: q rows r0..r0+15 (ISA 16-bit A 16x32 layout).
    // lane<16 : M=l16, halves = K 0-7  then K 16-23
    // lane>=16: M=l16, halves = K 8-15 then K 24-31
    ABFrag afrag;
    {
        const _Float16* qr = qbuf + ((size_t)b * N_ + r0 + l16) * QK_;
        afrag.f4[0] = *(const v4f*)(qr + half * 8);
        afrag.f4[1] = *(const v4f*)(qr + 16 + half * 8);
    }

    float accs[8];
#pragma unroll
    for (int g = 0; g < 8; ++g) accs[g] = 0.0f;

    // Per-lane G base: rows r0 + half*8 + g, column (ct + l16).
    const float* Gb = G + (size_t)b * N_ * N_ + (size_t)(r0 + half * 8) * N_ + l16;

    for (int tcol = 0; tcol < 16; ++tcol) {
        const int ct = wave * 256 + tcol * 16;

        // B fragment: k columns ct..ct+15 (lane<16: K 0-7 then 16-23; lane>=16: K 8-15 then 24-31).
        ABFrag bfrag;
        const _Float16* kr = kbuf + ((size_t)b * N_ + ct + l16) * QK_;
        bfrag.f4[0] = *(const v4f*)(kr + half * 8);
        bfrag.f4[1] = *(const v4f*)(kr + 16 + half * 8);

        v8f c = {};
        c = __builtin_amdgcn_wmma_f32_16x16x32_f16(false, afrag.v, false, bfrag.v,
                                                   (short)0, c, false, false);

        const float* gp = Gb + ct;
#pragma unroll
        for (int g = 0; g < 8; ++g) {
            float gval = __builtin_nontemporal_load(gp + (size_t)g * N_); // streamed once
            float wv = c[g] * c[g] * gval;
            accs[g] += wv;
            // C layout: VGPR g, lane<16 -> row g, lane>=16 -> row g+8; col = ct+l16.
            wtile[(size_t)(half * 8 + g) * N_ + ct + l16] = wv;
        }
    }

    // Row-sum reduction across each 16-lane half (rows g / g+8), then LDS atomics.
#pragma unroll
    for (int g = 0; g < 8; ++g) {
        float v = accs[g];
        v += __shfl_xor(v, 1, 32);
        v += __shfl_xor(v, 2, 32);
        v += __shfl_xor(v, 4, 32);
        v += __shfl_xor(v, 8, 32);
        if (l16 == 0) atomicAdd(&rowsum[half * 8 + g], v);
    }
    __syncthreads();

    if (tid < 16) inv[tid] = 1.0f / (rowsum[tid] + 1e-6f);
    __syncthreads();

    // Normalized write-out: fully coalesced 16-byte stores, non-temporal.
    float* ob = out + ((size_t)b * N_ + r0) * N_;
    for (int i = tid; i < 16 * N_ / 4; i += (int)blockDim.x) {
        const int row = (i * 4) >> 11;                 // / N_
        v4f v = *(const v4f*)(wtile + (size_t)i * 4);
        const float sc = inv[row];
        v *= sc;
        __builtin_nontemporal_store(v, (v4f*)(ob + (size_t)i * 4));
    }
}

// ---------------------------------------------------------------------------
extern "C" void kernel_launch(void* const* d_in, const int* in_sizes, int n_in,
                              void* d_out, int out_size, void* d_ws, size_t ws_size,
                              hipStream_t stream) {
    (void)in_sizes; (void)n_in; (void)out_size; (void)ws_size;
    const float* s  = (const float*)d_in[0];   // [8, 2048, 10]
    const float* G  = (const float*)d_in[1];   // [8, 2048, 2048]
    const float* Wq = (const float*)d_in[2];   // [32, 10]
    const float* Wk = (const float*)d_in[3];   // [32, 10]
    float* out = (float*)d_out;                // [8, 2048, 2048]

    _Float16* qbuf = (_Float16*)d_ws;                       // 1 MB
    _Float16* kbuf = qbuf + (size_t)B_ * N_ * QK_;          // 1 MB

    qk_project_kernel<<<(B_ * N_) / 256, 256, 0, stream>>>(s, Wq, Wk, qbuf, kbuf);

    const size_t ldsBytes = (size_t)(16 * N_ + 32) * sizeof(float); // 131200 B
    attn_row_kernel<<<B_ * (N_ / 16), 256, ldsBytes, stream>>>(qbuf, kbuf, G, out);
}
